// CausalSelfAttention_47021301957161
// MI455X (gfx1250) — matrix operand order repaired and checked by reading, not compiled, not run on hardware
//
#include <hip/hip_runtime.h>

#define BB  2
#define TT  2048
#define CC  2048
#define NHH 16
#define HSS 128
#define C3  6144

typedef __attribute__((ext_vector_type(16))) __bf16 v16bf;
typedef __attribute__((ext_vector_type(8)))  float  v8f;
typedef unsigned short u16;
typedef int vi4 __attribute__((vector_size(16)));
typedef __attribute__((address_space(1))) vi4* gvi4p;
typedef __attribute__((address_space(3))) vi4* lvi4p;

union Frag { v16bf v; int4 i[2]; };

#if __has_builtin(__builtin_amdgcn_global_load_async_to_lds_b128) && \
    __has_builtin(__builtin_amdgcn_s_wait_asynccnt)
#define USE_ASYNC_LDS 1
#else
#define USE_ASYNC_LDS 0
#endif

// 16-byte global -> LDS copy; async (ASYNCcnt) when the toolchain exposes it.
__device__ __forceinline__ void copy16_g2l(const u16* g, u16* l) {
#if USE_ASYNC_LDS
  __builtin_amdgcn_global_load_async_to_lds_b128(
      (gvi4p)(void*)(g), (lvi4p)(void*)(l), 0, 0);
#else
  *(int4*)l = *(const int4*)g;
#endif
}
__device__ __forceinline__ void async_wait0() {
#if USE_ASYNC_LDS
  __builtin_amdgcn_s_wait_asynccnt(0);
#endif
}

__device__ __forceinline__ v8f wmma_bf16(v16bf a, v16bf b, v8f c) {
  // (neg_a, A, neg_b, B, c_mod, C, reuse_a, reuse_b)
  return __builtin_amdgcn_wmma_f32_16x16x32_bf16(false, a, false, b, (short)0, c,
                                                 false, false);
}

__device__ __forceinline__ u16 f2bf(float f) {
  unsigned int u = __float_as_uint(f);
  u += 0x7FFFu + ((u >> 16) & 1u);   // round-to-nearest-even
  return (u16)(u >> 16);
}

__device__ __forceinline__ v8f vzero() {
  v8f z;
  #pragma unroll
  for (int i = 0; i < 8; ++i) z[i] = 0.0f;
  return z;
}

// ---------------------------------------------------------------- convert
__global__ void cvt_f32_to_bf16(const float* __restrict__ in,
                                u16* __restrict__ out, int n4) {
  int i = blockIdx.x * blockDim.x + threadIdx.x;
  if (i < n4) {
    float4 v = ((const float4*)in)[i];
    ushort4 o;
    o.x = f2bf(v.x); o.y = f2bf(v.y); o.z = f2bf(v.z); o.w = f2bf(v.w);
    ((ushort4*)out)[i] = o;
  }
}

// ---------------------------------------------------------------- GEMM
// out[M,N] = A[M,K] * W[N,K]^T   (A, W bf16 row-major; out bf16 or f32)
// 256 threads (8 waves); block tile 64(M) x 128(N); K double-buffered 32-wide.
template <bool BF16OUT>
__global__ __launch_bounds__(256) void gemm_bf16_wmma(
    const u16* __restrict__ A, const u16* __restrict__ W,
    void* __restrict__ outp, int M, int N, int K) {
  __shared__ u16 As[2][64 * 32];    // 2 x 4 KB
  __shared__ u16 Bs[2][128 * 32];   // 2 x 8 KB
  const int tid  = threadIdx.x;
  const int lane = tid & 31, wid = tid >> 5;
  const int ln = lane & 15, hi = lane >> 4;
  const int wm = wid & 3, wn = wid >> 2;      // 4x2 wave grid
  const int m0 = blockIdx.y * 64;
  const int n0 = blockIdx.x * 128;

  v8f acc[4];
  #pragma unroll
  for (int t = 0; t < 4; ++t) acc[t] = vzero();

  const int arow = tid >> 2, aseg = tid & 3;  // 64 rows x 4 segs of 8 elems
  auto stage = [&](int buf, int k0) {
    copy16_g2l(&A[(m0 + arow) * K + k0 + aseg * 8],
               &As[buf][arow * 32 + aseg * 8]);
    #pragma unroll
    for (int half = 0; half < 2; ++half) {
      int r = arow + half * 64;
      copy16_g2l(&W[(n0 + r) * K + k0 + aseg * 8],
                 &Bs[buf][r * 32 + aseg * 8]);
    }
  };

  stage(0, 0);
  async_wait0();
  __syncthreads();

  int cur = 0;
  for (int k0 = 0; k0 < K; k0 += 32) {
    if (k0 + 32 < K) stage(cur ^ 1, k0 + 32);   // overlap copy with compute
    if (k0 + 64 < K)
      __builtin_prefetch(&A[(m0 + arow) * K + k0 + 64 + aseg * 8], 0, 0);

    Frag af;  // A-layout: row ln, K chunks hi*8 and hi*8+16
    af.i[0] = *(const int4*)&As[cur][(wm * 16 + ln) * 32 + hi * 8];
    af.i[1] = *(const int4*)&As[cur][(wm * 16 + ln) * 32 + hi * 8 + 16];
    #pragma unroll
    for (int nt = 0; nt < 4; ++nt) {
      Frag bfr;  // B-layout: col ln, K chunks hi*8 and hi*8+16
      int nr = wn * 64 + nt * 16 + ln;
      bfr.i[0] = *(const int4*)&Bs[cur][nr * 32 + hi * 8];
      bfr.i[1] = *(const int4*)&Bs[cur][nr * 32 + hi * 8 + 16];
      acc[nt] = wmma_bf16(af.v, bfr.v, acc[nt]);
    }

    async_wait0();
    __syncthreads();
    cur ^= 1;
  }

  #pragma unroll
  for (int nt = 0; nt < 4; ++nt) {
    int col = n0 + wn * 64 + nt * 16 + ln;
    #pragma unroll
    for (int r = 0; r < 8; ++r) {
      int row = m0 + wm * 16 + hi * 8 + r;
      if (BF16OUT) ((u16*)outp)[row * N + col] = f2bf(acc[nt][r]);
      else         ((float*)outp)[row * N + col] = acc[nt][r];
    }
  }
}

// ---------------------------------------------------------------- flash attention
// qkv: bf16 [B, T, 3C]; y: bf16 [B, T, C].  128 threads (4 waves),
// each wave = one 16-row Q tile; block covers 64 query rows of one (b,h).
// Key columns inside a 32-key block are stored permuted (k -> 2k / 2k+1
// interleave) consistently in P and V^T; PV contraction is order-invariant.
__global__ __launch_bounds__(128) void flash_attn_wmma(
    const u16* __restrict__ qkv, u16* __restrict__ y) {
  __shared__ u16 Ks[32 * 128];     // 8 KB, row-major [key][d]
  __shared__ u16 Vt[128 * 32];     // 8 KB, transposed [d][permuted key]
  __shared__ u16 Ps[4][16 * 32];   // 4 KB, per-wave P tile [row][permuted key]
  const int tid  = threadIdx.x;
  const int lane = tid & 31, wv = tid >> 5;
  const int ln = lane & 15, hi = lane >> 4;
  const int qb = blockIdx.x, h = blockIdx.y, b = blockIdx.z;
  const int q0 = qb * 64;
  const u16* base = qkv + b * (TT * C3);

  Frag qf[4];  // Q 16x128 in A-layout, 4 chunks of K=32
  {
    const u16* qp = base + (q0 + wv * 16 + ln) * C3 + h * HSS;
    #pragma unroll
    for (int d = 0; d < 4; ++d) {
      qf[d].i[0] = *(const int4*)(qp + d * 32 + hi * 8);
      qf[d].i[1] = *(const int4*)(qp + d * 32 + hi * 8 + 16);
    }
  }

  v8f o[8];
  #pragma unroll
  for (int t = 0; t < 8; ++t) o[t] = vzero();
  float mrow[8], lrow[8];
  #pragma unroll
  for (int r = 0; r < 8; ++r) { mrow[r] = -3.0e38f; lrow[r] = 0.0f; }

  const float scale = 0.08838834764831845f;  // 1/sqrt(128)
  const int nkb = q0 / 32 + 2;               // causal: keys [0, q0+64)
  const int skey = tid >> 2, sseg = tid & 3;
  const int pcol = (skey < 16) ? (2 * skey) : (2 * (skey - 16) + 1);

  for (int kb = 0; kb < nkb; ++kb) {
    const int key0 = kb * 32;
    __syncthreads();
    {  // stage K (async, row-major) and V (manual transpose, permuted cols)
      const u16* kp = base + (key0 + skey) * C3 + CC + h * HSS + sseg * 32;
      const u16* vp = kp + CC;
      #pragma unroll
      for (int c = 0; c < 4; ++c) {
        copy16_g2l(kp + c * 8, &Ks[skey * 128 + sseg * 32 + c * 8]);
        int4 vv = *(const int4*)(vp + c * 8);
        const u16* ve = (const u16*)&vv;
        #pragma unroll
        for (int j = 0; j < 8; ++j)
          Vt[(sseg * 32 + c * 8 + j) * 32 + pcol] = ve[j];
      }
    }
    async_wait0();
    __syncthreads();

    // S = Q * K^T : two 16-key tiles, K-dim = d (4 chunks of 32)
    v8f s[2];
    #pragma unroll
    for (int nt = 0; nt < 2; ++nt) {
      v8f a = vzero();
      #pragma unroll
      for (int d = 0; d < 4; ++d) {
        Frag kf;
        kf.i[0] = *(const int4*)&Ks[(nt * 16 + ln) * 128 + d * 32 + hi * 8];
        kf.i[1] = *(const int4*)&Ks[(nt * 16 + ln) * 128 + d * 32 + hi * 8 + 16];
        a = wmma_bf16(qf[d].v, kf.v, a);
      }
      s[nt] = a;
    }

    // online softmax; lanes 0-15 hold rows hi*8+r with hi=0, 16-31 with hi=1
    #pragma unroll
    for (int r = 0; r < 8; ++r) {
      const int rowg = q0 + wv * 16 + hi * 8 + r;
      float s0 = s[0][r] * scale;
      float s1 = s[1][r] * scale;
      if (key0 + ln > rowg)      s0 = -3.0e38f;
      if (key0 + 16 + ln > rowg) s1 = -3.0e38f;
      float t = fmaxf(s0, s1);
      #pragma unroll
      for (int msk = 8; msk >= 1; msk >>= 1)
        t = fmaxf(t, __shfl_xor(t, msk, 16));
      const float mn    = fmaxf(mrow[r], t);
      const float alpha = __expf(mrow[r] - mn);
      const float p0 = __expf(s0 - mn);
      const float p1 = __expf(s1 - mn);
      float rs = p0 + p1;
      #pragma unroll
      for (int msk = 8; msk >= 1; msk >>= 1)
        rs += __shfl_xor(rs, msk, 16);
      lrow[r] = lrow[r] * alpha + rs;
      mrow[r] = mn;
      #pragma unroll
      for (int dt = 0; dt < 8; ++dt) o[dt][r] *= alpha;
      // keys ln (tile0) and 16+ln (tile1) land at permuted cols 2ln, 2ln+1:
      const int prow = hi * 8 + r;
      unsigned int pk = (unsigned int)f2bf(p0) | ((unsigned int)f2bf(p1) << 16);
      *(unsigned int*)&Ps[wv][prow * 32 + 2 * ln] = pk;
    }
    __syncthreads();

    // O += P * V  (A = P 16x32 keys, B = V 32x16 per d-tile)
    Frag pf;
    pf.i[0] = *(const int4*)&Ps[wv][ln * 32 + hi * 8];
    pf.i[1] = *(const int4*)&Ps[wv][ln * 32 + hi * 8 + 16];
    #pragma unroll
    for (int dt = 0; dt < 8; ++dt) {
      Frag vf;
      vf.i[0] = *(const int4*)&Vt[(dt * 16 + ln) * 32 + hi * 8];
      vf.i[1] = *(const int4*)&Vt[(dt * 16 + ln) * 32 + hi * 8 + 16];
      o[dt] = wmma_bf16(pf.v, vf.v, o[dt]);
    }
  }

  #pragma unroll
  for (int dt = 0; dt < 8; ++dt) {
    #pragma unroll
    for (int r = 0; r < 8; ++r) {
      const int rowg = q0 + wv * 16 + hi * 8 + r;
      const float val = o[dt][r] / lrow[r];
      y[(b * TT + rowg) * CC + h * HSS + dt * 16 + ln] = f2bf(val);
    }
  }
}

// ---------------------------------------------------------------- launch
extern "C" void kernel_launch(void* const* d_in, const int* in_sizes, int n_in,
                              void* d_out, int out_size, void* d_ws, size_t ws_size,
                              hipStream_t stream) {
  (void)in_sizes; (void)n_in; (void)out_size; (void)ws_size;
  const float* x  = (const float*)d_in[0];
  const float* wa = (const float*)d_in[1];
  const float* wp = (const float*)d_in[2];

  char* ws = (char*)d_ws;
  const size_t XB = 0;
  const size_t WA = XB + (size_t)BB * TT * CC * 2;    // x bf16      16 MB
  const size_t WP = WA + (size_t)3 * CC * CC * 2;     // w_attn bf16 24 MB
  const size_t QK = WP + (size_t)CC * CC * 2;         // w_proj bf16  8 MB
  const size_t YB = QK + (size_t)BB * TT * C3 * 2;    // qkv bf16    48 MB
  u16* xb   = (u16*)(ws + XB);
  u16* wab  = (u16*)(ws + WA);
  u16* wpb  = (u16*)(ws + WP);
  u16* qkvb = (u16*)(ws + QK);
  u16* yb   = (u16*)(ws + YB);                        // y bf16      16 MB

  const int nx  = BB * TT * CC / 4;
  const int nwa = 3 * CC * CC / 4;
  const int nwp = CC * CC / 4;
  cvt_f32_to_bf16<<<(nx  + 255) / 256, 256, 0, stream>>>(x,  xb,  nx);
  cvt_f32_to_bf16<<<(nwa + 255) / 256, 256, 0, stream>>>(wa, wab, nwa);
  cvt_f32_to_bf16<<<(nwp + 255) / 256, 256, 0, stream>>>(wp, wpb, nwp);

  // qkv = x @ w_attn^T : [4096 x 2048] * [6144 x 2048]^T -> bf16
  gemm_bf16_wmma<true><<<dim3(C3 / 128, BB * TT / 64), 256, 0, stream>>>(
      xb, wab, (void*)qkvb, BB * TT, C3, CC);

  // causal attention -> y bf16
  flash_attn_wmma<<<dim3(TT / 64, NHH, BB), 128, 0, stream>>>(qkvb, yb);

  // out = y @ w_proj^T : [4096 x 2048] * [2048 x 2048]^T -> f32
  gemm_bf16_wmma<false><<<dim3(CC / 128, BB * TT / 64), 256, 0, stream>>>(
      yb, wpb, d_out, BB * TT, CC, CC);
}
